// FastSequentialMNISTModel_86474871538193
// MI455X (gfx1250) — matrix operand order repaired and checked by reading, not compiled, not run on hardware
//
#include <hip/hip_runtime.h>

// ---------------------------------------------------------------------------
// Dendritic 2-cell RNN (B=8192, T=784) as a persistent WMMA kernel.
// One workgroup (8 waves, wave32) owns a 16-row batch tile for all 784 steps.
// All GEMMs use v_wmma_f32_16x16x32_f16. Weights repacked fp32->f16 once into
// d_ws and streamed from L2 every timestep (anti-hoist asm + addrspace(1)
// pointers => global_load_b128, LOADcnt-only). States s1/s2 in WMMA-C regs.
// ---------------------------------------------------------------------------

typedef _Float16 h8   __attribute__((ext_vector_type(8)));
typedef _Float16 v16h __attribute__((ext_vector_type(16)));
typedef float    v8f  __attribute__((ext_vector_type(8)));

// Global-address-space views of the f16 weight arena: guarantees
// global_load_b128 (LOADcnt only) instead of flat_load (LOADcnt+DScnt).
typedef __attribute__((address_space(1))) const _Float16 gf16;
typedef __attribute__((address_space(1))) const h8      gh8;

#define T_STEPS 784
#define TAIL    50
#define K1P     96      // 1 + 64 padded to 3*32
#define C1      256     // H1*NB
#define K2      192     // H1 + H2 = 6*32
#define C2      512     // H2*NB
#define KO      128     // H2
#define NOUT    10

// f16 weight arena offsets (elements) inside d_ws
#define OFF_W1B  0
#define OFF_W1O  (OFF_W1B + 256*K1P)
#define OFF_W2B  (OFF_W1O + 64*C1)
#define OFF_W2O  (OFF_W2B + 512*K2)
#define OFF_WOUT (OFF_W2O + 128*C2)
#define WS_ELEMS (OFF_WOUT + 16*KO)

// A-matrix 16x32 f16 tile load from LDS (ISA 7.12.2): lane<16 holds K 0..7 &
// 16..23, lane>=16 holds K 8..15 & 24..31, row M = lane&15.
__device__ __forceinline__ v16h load_a(const _Float16* base, int ld, int kbase,
                                       int r16, int hi) {
  const _Float16* p = base + r16 * ld + kbase + hi * 8;
  h8 lo = *(const h8*)(p);
  h8 hx = *(const h8*)(p + 16);
  return __builtin_shufflevector(lo, hx, 0,1,2,3,4,5,6,7,8,9,10,11,12,13,14,15);
}

// B-matrix 32x16 f16 tile load from global: lane<16 holds K 0..7 & 16..23 of
// column N=lane, lane>=16 holds K 8..15 & 24..31 of column N=lane-16. Weight stored
// [out x in] row-major, so a column of B == a contiguous row of W.
__device__ __forceinline__ v16h load_bg(gf16* base, int ld, int kbase,
                                        int r16, int hi) {
  gf16* p = base + r16 * ld + kbase + hi * 8;
  h8 lo = *(gh8*)(p);
  h8 hx = *(gh8*)(p + 16);
  return __builtin_shufflevector(lo, hx, 0,1,2,3,4,5,6,7,8,9,10,11,12,13,14,15);
}

__device__ __forceinline__ v8f wmma_f16(v16h a, v16h b, v8f c) {
  return __builtin_amdgcn_wmma_f32_16x16x32_f16(false, a, false, b,
                                                (short)0, c, false, false);
}

// ---------------------------------------------------------------------------
// One-time fp32 -> f16 weight repack with zero padding.
// ---------------------------------------------------------------------------
__global__ void prep_weights(const float* __restrict__ W1b,
                             const float* __restrict__ W1o,
                             const float* __restrict__ W2b,
                             const float* __restrict__ W2o,
                             const float* __restrict__ Wout,
                             _Float16* __restrict__ ws) {
  int i = blockIdx.x * 256 + threadIdx.x;
  if (i < 256 * K1P) {                       // W1b: 256 x 65 -> 256 x 96 (pad K)
    int r = i / K1P, c = i - r * K1P;
    ws[OFF_W1B + i] = (c < 65) ? (_Float16)W1b[r * 65 + c] : (_Float16)0.f;
    return;
  }
  i -= 256 * K1P;
  if (i < 64 * C1) { ws[OFF_W1O + i] = (_Float16)W1o[i]; return; }
  i -= 64 * C1;
  if (i < 512 * K2) { ws[OFF_W2B + i] = (_Float16)W2b[i]; return; }
  i -= 512 * K2;
  if (i < 128 * C2) { ws[OFF_W2O + i] = (_Float16)W2o[i]; return; }
  i -= 128 * C2;
  if (i < 16 * KO) {                         // Wout: 10 x 128 -> 16 x 128 (pad N)
    int r = i / KO, c = i - r * KO;
    ws[OFF_WOUT + i] = (r < NOUT) ? (_Float16)Wout[r * KO + c] : (_Float16)0.f;
  }
}

// ---------------------------------------------------------------------------
// Persistent recurrent kernel. gridDim.x = B/16, blockDim.x = 256.
// ---------------------------------------------------------------------------
__launch_bounds__(256)
__global__ void drnn_kernel(const float* __restrict__ x,
                            const float* __restrict__ b1b,
                            const float* __restrict__ b1o,
                            const float* __restrict__ b2b,
                            const float* __restrict__ b2o,
                            const float* __restrict__ bout,
                            const _Float16* __restrict__ w,
                            float* __restrict__ out) {
  __shared__ alignas(16) _Float16 comb1[16 * K1P];   // [x_t | h1 | pad]
  __shared__ alignas(16) _Float16 comb2[16 * K2];    // [h1 | h2]
  __shared__ alignas(16) _Float16 s1h[16 * C1];      // f16 staging of s1
  __shared__ alignas(16) _Float16 s2h[16 * C2];      // f16 staging of s2
  __shared__ alignas(16) _Float16 xh[16 * T_STEPS];  // whole x tile, f16

  const int tid  = threadIdx.x;
  const int wid  = tid >> 5;        // wave id 0..7
  const int lane = tid & 31;
  const int r16  = lane & 15;       // N index within a 16-wide tile
  const int hi   = lane >> 4;       // half-wave selector
  const long b0  = (long)blockIdx.x * 16;

  // --- init: zero recurrent LDS, pre-convert x tile to f16 ---
  for (int i = tid; i < 16 * K1P; i += 256) comb1[i] = (_Float16)0.f;
  for (int i = tid; i < 16 * K2; i += 256) comb2[i] = (_Float16)0.f;
  for (int i = tid; i < 16 * C1; i += 256) s1h[i] = (_Float16)0.f;
  for (int i = tid; i < 16 * C2; i += 256) s2h[i] = (_Float16)0.f;
  for (int i = tid; i < 16 * T_STEPS; i += 256) {
    int r = i / T_STEPS, t = i - r * T_STEPS;
    xh[i] = (_Float16)x[(b0 + r) * T_STEPS + t];
  }
  __syncthreads();
  if (tid < 16) comb1[tid * K1P] = xh[tid * T_STEPS];  // x_0 into column 0
  __syncthreads();

  // --- per-lane constants: alpha depends only on col%4 == lane&3 ---
  const float alpha = 0.1f + 0.2f * (float)(lane & 3);
  const float oma   = 1.0f - alpha;
  float bb1[2], bb2[4];
#pragma unroll
  for (int t = 0; t < 2; ++t) bb1[t] = b1b[wid * 32 + t * 16 + r16];
#pragma unroll
  for (int t = 0; t < 4; ++t) bb2[t] = b2b[wid * 64 + t * 16 + r16];
  const float bo1 = (wid < 4) ? b1o[wid * 16 + r16] : 0.f;
  const float bo2 = b2o[wid * 16 + r16];
  const float boo = (r16 < NOUT) ? bout[r16] : 0.f;

  v8f s1c[2] = {};   // cell1 states: 2 col-tiles/wave (cols wid*32..wid*32+31)
  v8f s2c[4] = {};   // cell2 states: 4 col-tiles/wave (cols wid*64..wid*64+63)
  v8f acc   = {};    // logits accumulator (wave 0 only)

  // Opaque copy of the weight base: refreshed through an asm barrier each
  // timestep so the compiler cannot hoist the (loop-invariant) weight tile
  // loads out of the t-loop and register-pin/spill them. Weights stream from
  // L2 (~414 KB total, shared by all workgroups) every step instead.
  unsigned long wp = (unsigned long)w;

  for (int t = 0; t < T_STEPS; ++t) {
    asm volatile("" : "+s"(wp));     // launder: weight loads stay in the loop
    gf16* W1b = (gf16*)wp + OFF_W1B;
    gf16* W1o = (gf16*)wp + OFF_W1O;
    gf16* W2b = (gf16*)wp + OFF_W2B;
    gf16* W2o = (gf16*)wp + OFF_W2O;
    gf16* Wou = (gf16*)wp + OFF_WOUT;

    // ---- Phase A: branch1 = comb1 @ W1b^T (K=96), state decay, stage f16 ----
#pragma unroll
    for (int tau = 0; tau < 2; ++tau) {
      const int n0 = wid * 32 + tau * 16;
      v8f c = {};
#pragma unroll
      for (int k = 0; k < 3; ++k)
        c = wmma_f16(load_a(comb1, K1P, k * 32, r16, hi),
                     load_bg(W1b + n0 * K1P, K1P, k * 32, r16, hi), c);
#pragma unroll
      for (int i = 0; i < 8; ++i) {
        float s = alpha * s1c[tau][i] + oma * (c[i] + bb1[tau]);
        s1c[tau][i] = s;
        s1h[(i + hi * 8) * C1 + n0 + r16] = (_Float16)s;
      }
    }
    __syncthreads();

    // ---- Phase B: out1 = s1 @ W1o^T (K=256), threshold -> h1 (waves 0..3) ----
    if (wid < 4) {
      const int n0 = wid * 16;
      v8f c0 = {}, c1 = {};                  // two independent K-chains
#pragma unroll
      for (int k = 0; k < 4; ++k) {
        c0 = wmma_f16(load_a(s1h, C1, k * 32, r16, hi),
                      load_bg(W1o + n0 * C1, C1, k * 32, r16, hi), c0);
        c1 = wmma_f16(load_a(s1h, C1, (k + 4) * 32, r16, hi),
                      load_bg(W1o + n0 * C1, C1, (k + 4) * 32, r16, hi), c1);
      }
      v8f c = c0 + c1;
#pragma unroll
      for (int i = 0; i < 8; ++i) {
        float o = c[i] + bo1;
        float sp = (o > 1.0f) ? o : 0.f;       // Threshold(1.0, 0.0)
        const int row = i + hi * 8;
        comb1[row * K1P + 1 + n0 + r16] = (_Float16)sp;  // h1 for next step
        comb2[row * K2 + n0 + r16]      = (_Float16)sp;  // h1 into cell2 input
      }
    }
    __syncthreads();

    // ---- Phase C: branch2 = comb2 @ W2b^T (K=192), decay, stage f16 ----
#pragma unroll
    for (int tau = 0; tau < 4; ++tau) {
      const int n0 = wid * 64 + tau * 16;
      v8f c = {};
#pragma unroll
      for (int k = 0; k < 6; ++k)
        c = wmma_f16(load_a(comb2, K2, k * 32, r16, hi),
                     load_bg(W2b + n0 * K2, K2, k * 32, r16, hi), c);
#pragma unroll
      for (int i = 0; i < 8; ++i) {
        float s = alpha * s2c[tau][i] + oma * (c[i] + bb2[tau]);
        s2c[tau][i] = s;
        s2h[(i + hi * 8) * C2 + n0 + r16] = (_Float16)s;
      }
    }
    __syncthreads();

    // ---- Phase D: out2 = s2 @ W2o^T (K=512), threshold -> h2 ----
    {
      const int n0 = wid * 16;
      v8f c0 = {}, c1 = {};                  // two independent K-chains
#pragma unroll
      for (int k = 0; k < 8; ++k) {
        c0 = wmma_f16(load_a(s2h, C2, k * 32, r16, hi),
                      load_bg(W2o + n0 * C2, C2, k * 32, r16, hi), c0);
        c1 = wmma_f16(load_a(s2h, C2, (k + 8) * 32, r16, hi),
                      load_bg(W2o + n0 * C2, C2, (k + 8) * 32, r16, hi), c1);
      }
      v8f c = c0 + c1;
#pragma unroll
      for (int i = 0; i < 8; ++i) {
        float o = c[i] + bo2;
        float sp = (o > 1.0f) ? o : 0.f;
        comb2[(i + hi * 8) * K2 + 64 + n0 + r16] = (_Float16)sp;  // h2
      }
    }
    __syncthreads();

    // ---- Phase E: logits accumulation over last TAIL steps (wave 0) ----
    if (t >= T_STEPS - TAIL && wid == 0) {
      v8f c = acc;
#pragma unroll
      for (int k = 0; k < 4; ++k)
        c = wmma_f16(load_a(comb2 + 64, K2, k * 32, r16, hi),
                     load_bg(Wou, KO, k * 32, r16, hi), c);
#pragma unroll
      for (int i = 0; i < 8; ++i) c[i] += boo;
      acc = c;
    }
    // stage x_{t+1} into comb1 column 0 (wave 1)
    if (wid == 1 && lane < 16 && (t + 1) < T_STEPS)
      comb1[lane * K1P] = xh[lane * T_STEPS + t + 1];
    __syncthreads();
  }

  // ---- final store: C layout -> out[b, n], n = r16 < 10, rows = vgpr+hi*8 ----
  if (wid == 0 && r16 < NOUT) {
#pragma unroll
    for (int i = 0; i < 8; ++i)
      out[(b0 + i + hi * 8) * NOUT + r16] = acc[i] * (1.0f / (float)TAIL);
  }
}

// ---------------------------------------------------------------------------
extern "C" void kernel_launch(void* const* d_in, const int* in_sizes, int n_in,
                              void* d_out, int out_size, void* d_ws,
                              size_t ws_size, hipStream_t stream) {
  const float* x    = (const float*)d_in[0];
  const float* W1b  = (const float*)d_in[1];
  const float* b1b  = (const float*)d_in[2];
  const float* W1o  = (const float*)d_in[3];
  const float* b1o  = (const float*)d_in[4];
  const float* W2b  = (const float*)d_in[5];
  const float* b2b  = (const float*)d_in[6];
  const float* W2o  = (const float*)d_in[7];
  const float* b2o  = (const float*)d_in[8];
  const float* Wout = (const float*)d_in[9];
  const float* bout = (const float*)d_in[10];

  _Float16* wsw = (_Float16*)d_ws;
  float* out = (float*)d_out;

  const int B = in_sizes[0] / T_STEPS;       // 8192
  const int prep_blocks = (WS_ELEMS + 255) / 256;

  prep_weights<<<prep_blocks, 256, 0, stream>>>(W1b, W1o, W2b, W2o, Wout, wsw);
  drnn_kernel<<<B / 16, 256, 0, stream>>>(x, b1b, b1o, b2b, b2o, bout, wsw, out);
}
